// TransformerEncoder_13434657702054
// MI455X (gfx1250) — matrix operand order repaired and checked by reading, not compiled, not run on hardware
//
#include <hip/hip_runtime.h>
#include <hip/hip_bf16.h>

// ---------------------------------------------------------------------------
// Types for CDNA5 WMMA (gfx1250, wave32)
// ---------------------------------------------------------------------------
typedef __attribute__((ext_vector_type(16))) __bf16 v16bf;
typedef __attribute__((ext_vector_type(8)))  float  v8f;
typedef __attribute__((ext_vector_type(4)))  int    v4i;

#define AS1 __attribute__((address_space(1)))
#define AS3 __attribute__((address_space(3)))

// CDNA5 async global->LDS copy path (ASYNCcnt-tracked), if this toolchain
// declares the builtins. Falls back to synchronous staging otherwise.
#if defined(__has_builtin)
#if __has_builtin(__builtin_amdgcn_global_load_async_to_lds_b128) && \
    __has_builtin(__builtin_amdgcn_s_wait_asynccnt)
#define USE_ASYNC_LDS 1
#endif
#endif

#ifdef USE_ASYNC_LDS
#pragma message("CDNA5: using global_load_async_to_lds_b128 for GEMM A-staging")
#else
#pragma message("CDNA5: async-to-LDS builtins unavailable; synchronous staging")
#endif

union BFrag {
    v16bf v;
    unsigned short u[16];
    unsigned int   w[8];
};

__device__ __forceinline__ unsigned short f2bf(float f) {
    unsigned int u = __float_as_uint(f);
    unsigned int r = (u + 0x7FFFu + ((u >> 16) & 1u)) >> 16;
    return (unsigned short)r;
}

__device__ __forceinline__ float gelu_tanh(float x) {
    const float c0 = 0.7978845608028654f;   // sqrt(2/pi)
    const float c1 = 0.044715f;
    float t = tanhf(c0 * (x + c1 * x * x * x));
    return 0.5f * x * (1.0f + t);
}

// ---------------------------------------------------------------------------
// fp32 -> bf16 conversion (weights)
// ---------------------------------------------------------------------------
__global__ __launch_bounds__(256)
void f2bf_kernel(const float* __restrict__ in, unsigned short* __restrict__ out,
                 size_t n) {
    size_t i = (size_t)blockIdx.x * blockDim.x + threadIdx.x;
    size_t stride = (size_t)gridDim.x * blockDim.x;
    for (; i < n; i += stride) out[i] = f2bf(in[i]);
}

// ---------------------------------------------------------------------------
// LayerNorm over D=512, one wave (32 lanes) per token, bf16 output
// ---------------------------------------------------------------------------
__global__ __launch_bounds__(256)
void ln_kernel(const float* __restrict__ x, const float* __restrict__ g,
               const float* __restrict__ bia, unsigned short* __restrict__ out,
               int ntok) {
    int tok  = (int)((blockIdx.x * blockDim.x + threadIdx.x) >> 5);
    int lane = threadIdx.x & 31;
    if (tok >= ntok) return;
    const float* xr = x + (size_t)tok * 512;
    float vals[16];
    float s = 0.f, s2 = 0.f;
#pragma unroll
    for (int e = 0; e < 16; ++e) {
        float v = xr[lane + 32 * e];
        vals[e] = v;
        s += v;
        s2 += v * v;
    }
#pragma unroll
    for (int off = 16; off >= 1; off >>= 1) {
        s  += __shfl_xor(s,  off);
        s2 += __shfl_xor(s2, off);
    }
    float mean = s * (1.0f / 512.0f);
    float var  = s2 * (1.0f / 512.0f) - mean * mean;
    float inv  = rsqrtf(var + 1e-5f);
    unsigned short* orow = out + (size_t)tok * 512;
#pragma unroll
    for (int e = 0; e < 16; ++e) {
        int idx = lane + 32 * e;
        orow[idx] = f2bf((vals[e] - mean) * inv * g[idx] + bia[idx]);
    }
}

// ---------------------------------------------------------------------------
// bf16 GEMM:  C[M,N] = A[M,K](bf16) * W[K,N](bf16) + bias
//
// Block: 256 threads = 8 waves (2 M-groups x 4 N-groups).
// Macro-tile 128(M) x 128(N); each wave owns 64x32 = 4x2 WMMA tiles, so each
// 32-wide K step issues 8 v_wmma_f32_16x16x32_bf16 from 6 fragment loads.
// K staged in 64-wide chunks (LDS 36KB, stride-72 padding -> conflict-free).
// A-panel staged via CDNA5 async global->LDS copies when available.
// ---------------------------------------------------------------------------
enum { FLAG_BF16OUT = 1, FLAG_GELU = 2, FLAG_RES = 4 };

constexpr int KC  = 64;       // K-chunk per LDS stage
constexpr int KCA = KC + 8;   // padded LDS stride (ushorts) for A tile
constexpr int KCB = KC + 8;   // padded LDS stride for transposed B tile

template <int FLAGS>
__global__ __launch_bounds__(256)
void gemm_bf16_kernel(const unsigned short* __restrict__ A,
                      const unsigned short* __restrict__ W,
                      const float* __restrict__ bias,
                      const float* __restrict__ resid,
                      float* __restrict__ outF,
                      unsigned short* __restrict__ outB,
                      int M, int N, int K) {
    __shared__ unsigned short As[128 * KCA];   // 18 KB
    __shared__ unsigned short Bs[128 * KCB];   // 18 KB (transposed: [n][k])

    const int tid  = threadIdx.x;
    const int wave = tid >> 5;
    const int lane = tid & 31;
    const int n0   = blockIdx.x * 128;
    const int m0   = blockIdx.y * 128;

    const int wm    = wave >> 2;          // 0..1   -> 64-row slab
    const int wn    = wave & 3;           // 0..3   -> 32-col slab
    const int half  = lane >> 4;
    const int l15   = lane & 15;
    const int mbase = wm * 64;
    const int nbase = wn * 32;

    v8f acc[4][2];
#pragma unroll
    for (int mi = 0; mi < 4; ++mi)
#pragma unroll
        for (int ni = 0; ni < 2; ++ni) acc[mi][ni] = (v8f){};

    const unsigned int* As32 = (const unsigned int*)As;
    const unsigned int* Bs32 = (const unsigned int*)Bs;

    for (int kc = 0; kc < K; kc += KC) {
        // ---- stage A tile: rows [m0,m0+128), cols [kc,kc+KC)
#ifdef USE_ASYNC_LDS
        {
#pragma unroll
            for (int it = 0; it < 4; ++it) {
                int lin = (it * 256 + tid) * 8;      // ushort index in 128x64 tile
                int row = lin >> 6;                  // / KC
                int col = lin & 63;                  // % KC
                void* gp = (void*)(A + (size_t)(m0 + row) * K + kc + col);
                void* lp = (void*)&As[row * KCA + col];
                __builtin_amdgcn_global_load_async_to_lds_b128(
                    (AS1 v4i*)gp, (AS3 v4i*)lp, 0, 0);
            }
        }
#else
        {
            const uint4* src = (const uint4*)(A + (size_t)m0 * K + kc);
#pragma unroll
            for (int it = 0; it < 4; ++it) {
                int lin = (it * 256 + tid) * 8;      // ushort index in 128x64 tile
                int row = lin >> 6;                  // / KC
                int col = lin & 63;                  // % KC
                uint4 d = src[(size_t)row * (K >> 3) + (col >> 3)];
                *(uint4*)&As[row * KCA + col] = d;
                if (kc + KC < K)
                    __builtin_prefetch(A + (size_t)(m0 + row) * K + kc + KC + col, 0, 1);
            }
        }
#endif
        // ---- stage B tile transposed: Bs[n*KCB + k] = W[(kc+k)*N + n0+n]
        //      packed dword along N for coalesced global reads
        {
#pragma unroll
            for (int it = 0; it < 16; ++it) {
                int lin = it * 256 + tid;            // 0..4095
                int k   = lin >> 6;                  // 0..63
                int n2  = (lin & 63) * 2;            // 0,2,...,126
                unsigned int d = *(const unsigned int*)&W[(size_t)(kc + k) * N + n0 + n2];
                Bs[n2 * KCB + k]       = (unsigned short)(d & 0xffffu);
                Bs[(n2 + 1) * KCB + k] = (unsigned short)(d >> 16);
            }
        }
#ifdef USE_ASYNC_LDS
        __builtin_amdgcn_s_wait_asynccnt(0);
#endif
        __syncthreads();

#pragma unroll
        for (int kk = 0; kk < KC; kk += 32) {
            BFrag a[4], b[2];
            // A fragments: lane = M row, regs hold K (lo-half K0..15, hi K16..31)
#pragma unroll
            for (int mi = 0; mi < 4; ++mi) {
                int abase = (mbase + mi * 16 + l15) * KCA + kk + half * 16;
#pragma unroll
                for (int v = 0; v < 4; ++v) a[mi].w[v]     = As32[(abase + 2 * v) >> 1];
#pragma unroll
                for (int v = 0; v < 4; ++v) a[mi].w[4 + v] = As32[(abase + 8 + 2 * v) >> 1];
            }
            // B fragments: lane = N col, regs hold K pairs
#pragma unroll
            for (int ni = 0; ni < 2; ++ni) {
                int bbase = (nbase + ni * 16 + l15) * KCB + kk + half * 16;
#pragma unroll
                for (int v = 0; v < 8; ++v) b[ni].w[v] = Bs32[(bbase + 2 * v) >> 1];
            }
            // 8 WMMAs per K step (A reused across N, B reused across M)
#pragma unroll
            for (int mi = 0; mi < 4; ++mi)
#pragma unroll
                for (int ni = 0; ni < 2; ++ni)
                    acc[mi][ni] = __builtin_amdgcn_wmma_f32_16x16x32_bf16(
                        false, a[mi].v, false, b[ni].v, (short)0, acc[mi][ni],
                        false, false);
        }
        __syncthreads();
    }

    // ---- epilogue: C layout lane=(N%16, half selects M+8), reg r = M row
#pragma unroll
    for (int ni = 0; ni < 2; ++ni) {
        const int col  = n0 + nbase + ni * 16 + l15;
        const float bc = bias[col];
#pragma unroll
        for (int mi = 0; mi < 4; ++mi) {
            const int rbase = m0 + mbase + mi * 16 + half * 8;
#pragma unroll
            for (int r = 0; r < 8; ++r) {
                int row   = rbase + r;
                float val = acc[mi][ni][r] + bc;
                if (FLAGS & FLAG_GELU) val = gelu_tanh(val);
                if (FLAGS & FLAG_RES) {
                    val += resid[(size_t)row * N + col];
                    outF[(size_t)row * N + col] = val;
                }
                if (FLAGS & FLAG_BF16OUT) outB[(size_t)row * N + col] = f2bf(val);
            }
        }
    }
}

// ---------------------------------------------------------------------------
// Shifted-window attention: one wave per (window, head).
//   grid = (512 windows, 4 head-groups), block = 128 (4 waves = 4 heads)
// Roll(-4,-4) + window partition folded into token addressing; shift mask is
// recomputed analytically from window coordinates (Swin region codes).
// ---------------------------------------------------------------------------
__global__ __launch_bounds__(128)
void attn_kernel(const unsigned short* __restrict__ qb,     // [32768,512] bf16
                 const unsigned short* __restrict__ kvb,    // [32768,1024] bf16 (k|v)
                 const int* __restrict__ rel_index,         // [64,64]
                 const float* __restrict__ rel_table,       // [225,16] this layer
                 unsigned short* __restrict__ ao) {         // [32768,512] bf16
    __shared__ unsigned short probsAll[4][64 * 72];         // padded vs bank conflicts

    const int wave = threadIdx.x >> 5;
    const int lane = threadIdx.x & 31;
    const int head = blockIdx.y * 4 + wave;
    const int win  = blockIdx.x;            // b*64 + wh*8 + ww
    const int b    = win >> 6;
    const int wh   = (win >> 3) & 7;
    const int ww   = win & 7;
    unsigned short* probs = probsAll[wave];

    auto tok_of = [&](int p) -> int {       // window pos -> source token (roll -4)
        int i  = p >> 3, j = p & 7;
        int hh = (wh * 8 + i + 4) & 63;
        int wc = (ww * 8 + j + 4) & 63;
        return ((b << 6) + hh) * 64 + wc;
    };
    auto code_of = [&](int p) -> int {      // Swin shift-mask region code
        int gh = wh * 8 + (p >> 3);
        int gw = ww * 8 + (p & 7);
        int rh = gh < 56 ? 0 : (gh < 60 ? 1 : 2);
        int rw = gw < 56 ? 0 : (gw < 60 ? 1 : 2);
        return rh * 3 + rw;
    };

    const unsigned int* qb32  = (const unsigned int*)qb;
    const unsigned int* kvb32 = (const unsigned int*)kvb;
    const int half = lane >> 4;
    const int l15  = lane & 15;

    // ---- K fragments (B operand of Q*K^T): lane = key%16, regs = dk
    BFrag kf[4];
#pragma unroll
    for (int nj = 0; nj < 4; ++nj) {
        int key  = nj * 16 + l15;
        int tok  = tok_of(key);
        int base = tok * 1024 + head * 32 + half * 16;   // ushort index
#pragma unroll
        for (int v = 0; v < 8; ++v) kf[nj].w[v] = kvb32[(base + 2 * v) >> 1];
    }

    const float scale = 0.17677669529663687f;  // 1/sqrt(32)

#pragma unroll
    for (int mi = 0; mi < 4; ++mi) {
        // Q fragment: lane = query row, regs = dk (K=32 matches dk exactly)
        BFrag qf;
        {
            int qpos = mi * 16 + l15;
            int tok  = tok_of(qpos);
            int base = tok * 512 + head * 32 + half * 16;
#pragma unroll
            for (int v = 0; v < 4; ++v) qf.w[v]     = qb32[(base + 2 * v) >> 1];
#pragma unroll
            for (int v = 0; v < 4; ++v) qf.w[4 + v] = qb32[(base + 8 + 2 * v) >> 1];
        }
        v8f s[4];
#pragma unroll
        for (int nj = 0; nj < 4; ++nj) {
            v8f z = {};
            s[nj] = __builtin_amdgcn_wmma_f32_16x16x32_bf16(
                false, qf.v, false, kf[nj].v, (short)0, z, false, false);
        }
        // bias + mask
#pragma unroll
        for (int nj = 0; nj < 4; ++nj) {
            int kpos = nj * 16 + l15;
            int ck   = code_of(kpos);
#pragma unroll
            for (int r = 0; r < 8; ++r) {
                int qpos = mi * 16 + half * 8 + r;
                float v = s[nj][r] * scale +
                          rel_table[rel_index[qpos * 64 + kpos] * 16 + head];
                if (code_of(qpos) != ck) v = -1.0e9f;
                s[nj][r] = v;
            }
        }
        // softmax over 64 keys: each row lives in one 16-lane half x 4 frags
#pragma unroll
        for (int r = 0; r < 8; ++r) {
            float mx = fmaxf(fmaxf(s[0][r], s[1][r]), fmaxf(s[2][r], s[3][r]));
#pragma unroll
            for (int off = 8; off >= 1; off >>= 1) mx = fmaxf(mx, __shfl_xor(mx, off));
            float e0 = __expf(s[0][r] - mx);
            float e1 = __expf(s[1][r] - mx);
            float e2 = __expf(s[2][r] - mx);
            float e3 = __expf(s[3][r] - mx);
            float sum = e0 + e1 + e2 + e3;
#pragma unroll
            for (int off = 8; off >= 1; off >>= 1) sum += __shfl_xor(sum, off);
            float inv = 1.0f / sum;
            s[0][r] = e0 * inv; s[1][r] = e1 * inv;
            s[2][r] = e2 * inv; s[3][r] = e3 * inv;
        }
        // probs -> LDS (row-major [q][key], padded stride 72)
#pragma unroll
        for (int nj = 0; nj < 4; ++nj) {
            int kpos = nj * 16 + l15;
#pragma unroll
            for (int r = 0; r < 8; ++r) {
                int qpos = mi * 16 + half * 8 + r;
                probs[qpos * 72 + kpos] = f2bf(s[nj][r]);
            }
        }
    }

    // ---- V fragments: B[k][n] = V[key][dk], dk = ni*16 + lane
    BFrag vf[2][2];
#pragma unroll
    for (int ks = 0; ks < 2; ++ks) {
#pragma unroll
        for (int ni = 0; ni < 2; ++ni) {
            int dk = ni * 16 + l15;
#pragma unroll
            for (int e = 0; e < 16; ++e) {
                int key = ks * 32 + half * 8 + (e & 7) + (e >> 3) * 16;
                int tok = tok_of(key);
                vf[ks][ni].u[e] = kvb[(size_t)tok * 1024 + 512 + head * 32 + dk];
            }
        }
    }

    const unsigned int* probs32 = (const unsigned int*)probs;

#pragma unroll
    for (int mi = 0; mi < 4; ++mi) {
        v8f o0 = {}, o1 = {};
#pragma unroll
        for (int ks = 0; ks < 2; ++ks) {
            BFrag a;                          // A = probs tile (16 x 32 keys)
            int qrow = mi * 16 + l15;
            int base = qrow * 72 + ks * 32 + half * 8;
#pragma unroll
            for (int v = 0; v < 4; ++v) a.w[v]     = probs32[(base + 2 * v) >> 1];
#pragma unroll
            for (int v = 0; v < 4; ++v) a.w[4 + v] = probs32[(base + 16 + 2 * v) >> 1];
            o0 = __builtin_amdgcn_wmma_f32_16x16x32_bf16(
                false, a.v, false, vf[ks][0].v, (short)0, o0, false, false);
            o1 = __builtin_amdgcn_wmma_f32_16x16x32_bf16(
                false, a.v, false, vf[ks][1].v, (short)0, o1, false, false);
        }
        // store O (merge + roll back is same token mapping)
#pragma unroll
        for (int r = 0; r < 8; ++r) {
            int qpos = mi * 16 + half * 8 + r;
            int tok  = tok_of(qpos);
            size_t o = (size_t)tok * 512 + head * 32;
            ao[o + l15]      = f2bf(o0[r]);
            ao[o + 16 + l15] = f2bf(o1[r]);
        }
    }
}

// ---------------------------------------------------------------------------
// Host orchestration
// ---------------------------------------------------------------------------
extern "C" void kernel_launch(void* const* d_in, const int* in_sizes, int n_in,
                              void* d_out, int out_size, void* d_ws, size_t ws_size,
                              hipStream_t stream) {
    (void)in_sizes; (void)n_in; (void)out_size; (void)ws_size;

    const float* x_in      = (const float*)d_in[0];
    // d_in[1] = mask (recomputed analytically in-kernel)
    const int*   rel_index = (const int*)d_in[2];
    const float* Wq        = (const float*)d_in[3];
    const float* bq        = (const float*)d_in[4];
    const float* Wkv       = (const float*)d_in[5];
    const float* bkv       = (const float*)d_in[6];
    const float* Wo        = (const float*)d_in[7];
    const float* bo        = (const float*)d_in[8];
    const float* rel_table = (const float*)d_in[9];
    const float* ln1_g     = (const float*)d_in[10];
    const float* ln1_b     = (const float*)d_in[11];
    const float* ln2_g     = (const float*)d_in[12];
    const float* ln2_b     = (const float*)d_in[13];
    const float* W1        = (const float*)d_in[14];
    const float* b1        = (const float*)d_in[15];
    const float* W2        = (const float*)d_in[16];
    const float* b2        = (const float*)d_in[17];

    constexpr int Mtok = 32768, D = 512, Dff = 2048, Lyr = 4;

    char* ws = (char*)d_ws;
    size_t off = 0;
    auto alloc = [&](size_t bytes) {
        size_t p = off;
        off += (bytes + 255) & ~(size_t)255;
        return p;
    };
    float*          xb     = (float*)(ws + alloc((size_t)Mtok * D * 4));
    unsigned short* xn     = (unsigned short*)(ws + alloc((size_t)Mtok * D * 2));
    unsigned short* qbuf   = (unsigned short*)(ws + alloc((size_t)Mtok * D * 2));
    unsigned short* kvbuf  = (unsigned short*)(ws + alloc((size_t)Mtok * 2 * D * 2));
    unsigned short* aobuf  = (unsigned short*)(ws + alloc((size_t)Mtok * D * 2));
    unsigned short* m1buf  = (unsigned short*)(ws + alloc((size_t)Mtok * Dff * 2));
    unsigned short* wq_bf  = (unsigned short*)(ws + alloc((size_t)Lyr * D * D * 2));
    unsigned short* wkv_bf = (unsigned short*)(ws + alloc((size_t)Lyr * D * 2 * D * 2));
    unsigned short* wo_bf  = (unsigned short*)(ws + alloc((size_t)Lyr * D * D * 2));
    unsigned short* w1_bf  = (unsigned short*)(ws + alloc((size_t)Lyr * D * Dff * 2));
    unsigned short* w2_bf  = (unsigned short*)(ws + alloc((size_t)Lyr * Dff * D * 2));

    // residual stream lives in fp32 in workspace
    (void)hipMemcpyAsync(xb, x_in, (size_t)Mtok * D * 4, hipMemcpyDeviceToDevice, stream);

    auto conv = [&](const float* s, unsigned short* d, size_t n) {
        int blocks = (int)((n + 256 * 8 - 1) / (256 * 8));
        if (blocks > 4096) blocks = 4096;
        f2bf_kernel<<<blocks, 256, 0, stream>>>(s, d, n);
    };
    conv(Wq,  wq_bf,  (size_t)Lyr * D * D);
    conv(Wkv, wkv_bf, (size_t)Lyr * D * 2 * D);
    conv(Wo,  wo_bf,  (size_t)Lyr * D * D);
    conv(W1,  w1_bf,  (size_t)Lyr * D * Dff);
    conv(W2,  w2_bf,  (size_t)Lyr * Dff * D);

    for (int l = 0; l < Lyr; ++l) {
        // xn = LN1(x)
        ln_kernel<<<Mtok / 8, 256, 0, stream>>>(xb, ln1_g + l * D, ln1_b + l * D, xn, Mtok);
        // q = xn @ Wq + bq
        gemm_bf16_kernel<FLAG_BF16OUT><<<dim3(D / 128, Mtok / 128), 256, 0, stream>>>(
            xn, wq_bf + (size_t)l * D * D, bq + l * D, nullptr, nullptr, qbuf, Mtok, D, D);
        // kv = xn @ Wkv + bkv   (k = cols 0..511, v = cols 512..1023)
        gemm_bf16_kernel<FLAG_BF16OUT><<<dim3(2 * D / 128, Mtok / 128), 256, 0, stream>>>(
            xn, wkv_bf + (size_t)l * D * 2 * D, bkv + l * 2 * D, nullptr, nullptr, kvbuf,
            Mtok, 2 * D, D);
        // windowed attention (roll/partition/mask/softmax/merge fused)
        attn_kernel<<<dim3(512, 4), 128, 0, stream>>>(
            qbuf, kvbuf, rel_index, rel_table + (size_t)l * 225 * 16, aobuf);
        // x = x + ao @ Wo + bo
        gemm_bf16_kernel<FLAG_RES><<<dim3(D / 128, Mtok / 128), 256, 0, stream>>>(
            aobuf, wo_bf + (size_t)l * D * D, bo + l * D, xb, xb, nullptr, Mtok, D, D);
        // h = LN2(x)
        ln_kernel<<<Mtok / 8, 256, 0, stream>>>(xb, ln2_g + l * D, ln2_b + l * D, xn, Mtok);
        // m1 = gelu(h @ W1 + b1)
        gemm_bf16_kernel<FLAG_BF16OUT | FLAG_GELU><<<dim3(Dff / 128, Mtok / 128), 256, 0, stream>>>(
            xn, w1_bf + (size_t)l * D * Dff, b1 + l * Dff, nullptr, nullptr, m1buf,
            Mtok, Dff, D);
        // x = x + m1 @ W2 + b2
        gemm_bf16_kernel<FLAG_RES><<<dim3(D / 128, Mtok / 128), 256, 0, stream>>>(
            m1buf, w2_bf + (size_t)l * Dff * D, b2 + l * D, xb, xb, nullptr, Mtok, D, Dff);
    }

    (void)hipMemcpyAsync(d_out, xb, (size_t)Mtok * D * 4, hipMemcpyDeviceToDevice, stream);
}
